// QGrouping_24790551232812
// MI455X (gfx1250) — matrix operand order repaired and checked by reading, not compiled, not run on hardware
//
#include <hip/hip_runtime.h>
#include <hip/hip_bf16.h>

typedef __attribute__((ext_vector_type(16))) _Float16 v16h;
typedef __attribute__((ext_vector_type(8)))  float    v8f;

#define NCHUNK 32
#define BDIM 256

// ---------- helpers ----------
__device__ __forceinline__ int lb(const int* a, int n, int v) {
    int lo = 0, hi = n;
    while (lo < hi) { int m = (lo + hi) >> 1; if (a[m] < v) lo = m + 1; else hi = m; }
    return lo;
}
// monotonic float<->uint mapping for atomicMax on floats
__device__ __forceinline__ unsigned encf(float f) {
    unsigned u = __float_as_uint(f);
    return (u & 0x80000000u) ? ~u : (u | 0x80000000u);
}
__device__ __forceinline__ float decf(unsigned k) {
    return (k & 0x80000000u) ? __uint_as_float(k & 0x7FFFFFFFu) : __uint_as_float(~k);
}

// ---------- kernel 0: fold Wq@Wk into combined 48x256 B matrix, emit f16 frags ----------
// Wc rows 0..31 = Wv, rows 32..39 = Wq@Wk (weights), rows 40..47 = 0 (pad).
// Frag layout matches WMMA f16 B 32x16 per-lane layout:
//   frag[((step*3+t)*32 + lane)*16 + h] = Wc[t*16 + (lane&15)][step*32 + (lane>=16?8:0) + (h<8?h:h+8)]
__global__ void __launch_bounds__(BDIM) prep_kernel(
    const float* __restrict__ Wk, const float* __restrict__ bk,
    const float* __restrict__ Wq, const float* __restrict__ bq,
    const float* __restrict__ Wv, const float* __restrict__ bv,
    float* __restrict__ bc, _Float16* __restrict__ frag,
    unsigned* __restrict__ mbits, float* __restrict__ sbuf, float* __restrict__ embs)
{
    __shared__ float Wc[48 * 256];   // 49152 B LDS (gfx1250 WGP has 320 KB)
    const int tid = threadIdx.x;

    for (int e = tid; e < 48 * 256; e += BDIM) {
        int r = e >> 8, c = e & 255;
        float v;
        if (r < 32) {
            v = Wv[r * 256 + c];
        } else if (r < 40) {
            int q = r - 32; float s = 0.f;
            for (int j = 0; j < 100; ++j) s += Wq[q * 100 + j] * Wk[j * 256 + c];
            v = s;
        } else v = 0.f;
        Wc[e] = v;
    }
    if (tid < 48) {
        float v;
        if (tid < 32) v = bv[tid];
        else if (tid < 40) {
            int q = tid - 32; float s = bq[q];
            for (int j = 0; j < 100; ++j) s += Wq[q * 100 + j] * bk[j];
            v = s;
        } else v = 0.f;
        bc[tid] = v;
    }
    // zero per-launch accumulators (ws is poisoned / carries stale data otherwise)
    for (int e = tid; e < 1024; e += BDIM) { mbits[e] = 0u; sbuf[e] = 0.f; }
    for (int e = tid; e < 32768; e += BDIM) embs[e] = 0.f;

    __syncthreads();

    for (int f = tid; f < 24 * 512; f += BDIM) {   // 8 steps * 3 tiles * 32 lanes * 16 halfs
        int lane = (f >> 4) & 31;
        int h    = f & 15;
        int st   = f >> 9;              // step*3 + t
        int step = st / 3, t = st % 3;
        int colB = t * 16 + (lane & 15);
        int k    = step * 32 + ((lane >> 4) ? 8 : 0) + (h < 8 ? h : h + 8);
        frag[f] = (_Float16)Wc[colB * 256 + k];
    }
}

// ---------- kernel 1: fused GEMM  [N,256] x [256,48] -> val[N,32] (f16) + weights[N,8] (f32) ----------
__global__ void __launch_bounds__(BDIM) fused_gemm_kernel(
    const float* __restrict__ x, const _Float16* __restrict__ frag,
    const float* __restrict__ bc, float* __restrict__ wOut,
    _Float16* __restrict__ valH, int ntiles)
{
    const int lane = threadIdx.x & 31;
    const int wave = threadIdx.x >> 5;
    const int tile = blockIdx.x * (BDIM / 32) + wave;
    if (tile >= ntiles) return;                       // wave-uniform guard (EXEC stays full)
    const int tbase = tile * 16;
    const int col = lane & 15;
    const int hi  = lane >> 4;                         // 0 or 1

    const float* xr = x + (size_t)(tbase + col) * 256; // A row for this lane (both half-waves share row)
    v8f acc0 = {}, acc1 = {}, acc2 = {};

#pragma unroll
    for (int step = 0; step < 8; ++step) {
        const int klo = step * 32 + hi * 8;
        // A frag per ISA 16-bit A 16x32 layout: halfs 0..7 = K[klo..klo+7], 8..15 = K[klo+16..klo+23]
        float4 f0 = *reinterpret_cast<const float4*>(xr + klo);
        float4 f1 = *reinterpret_cast<const float4*>(xr + klo + 4);
        float4 f2 = *reinterpret_cast<const float4*>(xr + klo + 16);
        float4 f3 = *reinterpret_cast<const float4*>(xr + klo + 20);
        float fl[16] = { f0.x, f0.y, f0.z, f0.w, f1.x, f1.y, f1.z, f1.w,
                         f2.x, f2.y, f2.z, f2.w, f3.x, f3.y, f3.z, f3.w };
        v16h a;
#pragma unroll
        for (int i = 0; i < 16; ++i) a[i] = (_Float16)fl[i];

        const _Float16* fb = frag + (size_t)step * 3 * 512 + (size_t)lane * 16;
        v16h b0 = *reinterpret_cast<const v16h*>(fb);
        v16h b1 = *reinterpret_cast<const v16h*>(fb + 512);
        v16h b2 = *reinterpret_cast<const v16h*>(fb + 1024);

        acc0 = __builtin_amdgcn_wmma_f32_16x16x32_f16(false, a, false, b0, (short)0, acc0, false, false);
        acc1 = __builtin_amdgcn_wmma_f32_16x16x32_f16(false, a, false, b1, (short)0, acc1, false, false);
        acc2 = __builtin_amdgcn_wmma_f32_16x16x32_f16(false, a, false, b2, (short)0, acc2, false, false);
    }

    const float b0c = bc[col], b1c = bc[16 + col], b2c = bc[32 + col];
#pragma unroll
    for (int r = 0; r < 8; ++r) {
        // C/D layout: VGPR r -> M = r + 8*hi, N = col
        const int row = tbase + r + hi * 8;
        valH[(size_t)row * 32 + col]      = (_Float16)(acc0[r] + b0c);
        valH[(size_t)row * 32 + 16 + col] = (_Float16)(acc1[r] + b1c);
        if (col < 8) wOut[(size_t)row * 8 + col] = acc2[r] + b2c;
    }
}

// ---------- kernel 2: per-segment column max (chunked, atomicMax on encoded uint) ----------
__global__ void __launch_bounds__(BDIM) seg_max_kernel(
    const int* __restrict__ batch, const float* __restrict__ w,
    unsigned* __restrict__ mbits, int N)
{
    __shared__ int se[2];
    __shared__ float red[BDIM * 8];
    const int tid = threadIdx.x, b = blockIdx.x, chunk = blockIdx.y;
    if (tid == 0) { se[0] = lb(batch, N, b); se[1] = lb(batch, N, b + 1); }
    __syncthreads();
    const int start = se[0], end = se[1];
    const int len = end - start;
    const int per = (len + NCHUNK - 1) / NCHUNK;
    const int r0 = start + chunk * per;
    const int r1 = min(r0 + per, end);

    float mx[8];
#pragma unroll
    for (int k = 0; k < 8; ++k) mx[k] = -3.4e38f;
    for (int r = r0 + tid; r < r1; r += BDIM)
#pragma unroll
        for (int k = 0; k < 8; ++k) mx[k] = fmaxf(mx[k], w[(size_t)r * 8 + k]);
#pragma unroll
    for (int k = 0; k < 8; ++k) red[tid * 8 + k] = mx[k];
    __syncthreads();
    if (tid < 8) {
        float m = -3.4e38f;
        for (int i = 0; i < BDIM; ++i) m = fmaxf(m, red[i * 8 + tid]);
        atomicMax(&mbits[b * 8 + tid], encf(m));
    }
}

// ---------- kernel 3: e=exp(w-m); accumulate sum(e) and sum(e*val) per segment ----------
__global__ void __launch_bounds__(BDIM) seg_accum_kernel(
    const int* __restrict__ batch, const float* __restrict__ w,
    const _Float16* __restrict__ valH, const unsigned* __restrict__ mbits,
    float* __restrict__ sbuf, float* __restrict__ embs,
    float* __restrict__ eout, int N)
{
    __shared__ int se[2];
    __shared__ float m_sh[8];
    __shared__ float e_s[64 * 8];
    __shared__ float v_s[64 * 32];
    const int tid = threadIdx.x, b = blockIdx.x, chunk = blockIdx.y;
    if (tid == 0) { se[0] = lb(batch, N, b); se[1] = lb(batch, N, b + 1); }
    if (tid < 8)  m_sh[tid] = decf(mbits[b * 8 + tid]);
    __syncthreads();
    const int start = se[0], end = se[1];
    const int len = end - start;
    const int per = (len + NCHUNK - 1) / NCHUNK;
    const int r0 = start + chunk * per;
    const int r1 = min(r0 + per, end);

    const int k = tid >> 5, v = tid & 31;   // each thread owns output cell (k,v)
    float acc = 0.f, sacc = 0.f;

    for (int base = r0; base < r1; base += 64) {
        const int rows = min(64, r1 - base);
        for (int ei = tid; ei < rows * 8; ei += BDIM) {
            const int rl = ei >> 3, kk = ei & 7;
            const size_t gi = (size_t)(base + rl) * 8 + kk;
            const float e = __expf(w[gi] - m_sh[kk]);
            e_s[rl * 8 + kk] = e;
            eout[gi] = e;                    // numerator; normalized by kernel 4
        }
        for (int vi = tid; vi < rows * 32; vi += BDIM) {
            const int rl = vi >> 5, vv = vi & 31;
            v_s[rl * 32 + vv] = (float)valH[(size_t)(base + rl) * 32 + vv];
        }
        __syncthreads();
        for (int r = 0; r < rows; ++r) acc += e_s[r * 8 + k] * v_s[r * 32 + v];
        if (tid < 8) for (int r = 0; r < rows; ++r) sacc += e_s[r * 8 + tid];
        __syncthreads();
    }
    atomicAdd(&embs[(size_t)b * 256 + tid], acc);
    if (tid < 8) atomicAdd(&sbuf[b * 8 + tid], sacc);
}

// ---------- kernel 4: norm_w = e / s[seg] ----------
__global__ void __launch_bounds__(BDIM) norm_kernel(
    const int* __restrict__ batch, const float* __restrict__ sbuf,
    float* __restrict__ eout, int N)
{
    const int idx = blockIdx.x * BDIM + threadIdx.x;
    if (idx >= N * 8) return;
    const int n = idx >> 3, kk = idx & 7;
    const int seg = batch[n];
    const float sv = sbuf[seg * 8 + kk];
    const float e = eout[idx];
    eout[idx] = (sv > 0.f) ? e / sv : 0.f;
}

// ---------- kernel 5: embs = embs_num / s ----------
__global__ void __launch_bounds__(BDIM) embs_fin_kernel(
    const float* __restrict__ sbuf, const float* __restrict__ embs,
    float* __restrict__ out)
{
    const int idx = blockIdx.x * BDIM + threadIdx.x;   // 32768 total
    const int b = idx >> 8, kk = (idx >> 5) & 7;
    const float sv = sbuf[b * 8 + kk];
    out[idx] = (sv > 0.f) ? embs[idx] / sv : 0.f;
}

extern "C" void kernel_launch(void* const* d_in, const int* in_sizes, int n_in,
                              void* d_out, int out_size, void* d_ws, size_t ws_size,
                              hipStream_t stream) {
    const float* x   = (const float*)d_in[0];
    const int*   bat = (const int*)  d_in[1];
    const float* Wk  = (const float*)d_in[2];
    const float* bk  = (const float*)d_in[3];
    const float* Wq  = (const float*)d_in[4];
    const float* bq  = (const float*)d_in[5];
    const float* Wv  = (const float*)d_in[6];
    const float* bv  = (const float*)d_in[7];
    const int N = in_sizes[1];

    float* out      = (float*)d_out;
    float* norm_out = out + 128 * 8 * 32;   // embs first, then norm_w [N,8]

    // workspace carve-up
    char* ws = (char*)d_ws;
    float*     bc    = (float*)     (ws);                               // 256 B
    _Float16*  frag  = (_Float16*)  (ws + 256);                         // 24576 B
    unsigned*  mbits = (unsigned*)  (ws + 256 + 24576);                 // 4096 B
    float*     sbuf  = (float*)     (ws + 256 + 24576 + 4096);          // 4096 B
    float*     embs  = (float*)     (ws + 256 + 24576 + 8192);          // 131072 B
    float*     wbuf  = (float*)     (ws + 256 + 24576 + 8192 + 131072); // N*8*4 B
    _Float16*  valH  = (_Float16*)  ((char*)wbuf + (size_t)N * 8 * 4);  // N*32*2 B

    prep_kernel<<<1, BDIM, 0, stream>>>(Wk, bk, Wq, bq, Wv, bv, bc, frag, mbits, sbuf, embs);

    const int ntiles = (N + 15) / 16;       // N = 500000 -> 31250, multiple of 16 rows
    fused_gemm_kernel<<<(ntiles + 7) / 8, BDIM, 0, stream>>>(x, frag, bc, wbuf, valH, ntiles);

    dim3 gseg(128, NCHUNK);
    seg_max_kernel<<<gseg, BDIM, 0, stream>>>(bat, wbuf, mbits, N);
    seg_accum_kernel<<<gseg, BDIM, 0, stream>>>(bat, wbuf, valH, mbits, sbuf, embs, norm_out, N);

    norm_kernel<<<(N * 8 + BDIM - 1) / BDIM, BDIM, 0, stream>>>(bat, sbuf, norm_out, N);
    embs_fin_kernel<<<32768 / BDIM, BDIM, 0, stream>>>(sbuf, embs, out);
}
